// transPPI_43714177139193
// MI455X (gfx1250) — matrix operand order repaired and checked by reading, not compiled, not run on hardware
//
#include <hip/hip_runtime.h>

// ---------------- problem constants ----------------
static constexpr int NR  = 20000;   // struct nodes / residues
static constexpr int NQK = 19998;   // kmer nodes
static constexpr int ESN = 300000;  // struct edges
static constexpr int EQN = 300000;  // seq edges

// ---------------- WMMA types ----------------
typedef __attribute__((ext_vector_type(16))) __bf16 v16bf;
typedef __attribute__((ext_vector_type(8)))  float  v8f;

union FragAB { v16bf v; uint4 q[2]; };
union FragC  { v8f v; float f[8]; };

__device__ __forceinline__ unsigned short f2bf(float f) {
  union { float f; unsigned u; } v; v.f = f;
  unsigned r = v.u + 0x7FFFu + ((v.u >> 16) & 1u);   // round-to-nearest-even
  return (unsigned short)(r >> 16);
}

// =====================================================================
// Generic bf16 WMMA GEMM:  C[M,Nc] = A[M,KD](bf16) * Bp(packed bf16) + bias
// Block = 256 threads = 8 waves; wave(wM 0..3, wN 0..1) computes 16x64.
// KD is a compile-time constant so every A/B access is an immediate
// offset from a fixed base pointer (no per-iteration address VALU).
// Bp packed per 16x32 (NxK) tile: idx = ((nt*KT+kt)*32 + lane)*16 + e,
//   value = B[kt*32 + (lane>>4)*16 + e][nt*16 + (lane&15)]   (contiguous k = 16h+e, same order as A)
// Out-of-range rows: A row index is clamped to 0 (loads stay in-bounds);
// the resulting garbage C rows are masked at store time (row-independent).
// =====================================================================
template<int LEAKY, int KD>
__global__ __launch_bounds__(256) void k_gemm_bf16(
    const unsigned short* __restrict__ A, const unsigned short* __restrict__ Bp,
    const float* __restrict__ bias, float* __restrict__ C,
    int M, int Nc)
{
  constexpr int KT = KD / 32;
  const int lane = threadIdx.x & 31;
  const int wid  = threadIdx.x >> 5;
  const int wM   = wid & 3;
  const int wN   = wid >> 2;
  const int m0   = blockIdx.x * 64 + wM * 16;
  const int n0   = blockIdx.y * 128 + wN * 64;
  const int hf   = lane >> 4;
  int row = m0 + (lane & 15);
  if (row >= M) row = 0;                       // clamp; masked at store

  FragC acc[4];
#pragma unroll
  for (int t = 0; t < 4; ++t)
#pragma unroll
    for (int r = 0; r < 8; ++r) acc[t].f[r] = 0.f;

  const unsigned short* __restrict__ Ab = A + (size_t)row * KD + hf * 16;
  const unsigned short* __restrict__ Bb = Bp + ((size_t)(n0 >> 4) * KT * 32 + lane) * 16;

#pragma unroll
  for (int kt = 0; kt < KT; ++kt) {
    FragAB a;
    a.q[0] = *(const uint4*)(Ab + kt * 32);        // K = kt*32 + hf*16 .. +7
    a.q[1] = *(const uint4*)(Ab + kt * 32 + 8);    // K = kt*32 + hf*16 + 8 .. +15
#pragma unroll
    for (int t = 0; t < 4; ++t) {
      const unsigned short* bb = Bb + ((size_t)t * KT + kt) * 512;
      FragAB b;
      b.q[0] = *(const uint4*)(bb);
      b.q[1] = *(const uint4*)(bb + 8);
      acc[t].v = __builtin_amdgcn_wmma_f32_16x16x32_bf16(
          false, a.v, false, b.v, (short)0, acc[t].v, false, false);
    }
  }

  const int cn = lane & 15;
#pragma unroll
  for (int t = 0; t < 4; ++t) {
    const int ncol = n0 + t * 16 + cn;
    const float bv = bias[ncol];
#pragma unroll
    for (int r = 0; r < 8; ++r) {
      const int mr = m0 + hf * 8 + r;            // C layout: VGPR r -> M = hf*8+r
      if (mr < M) {
        float v = acc[t].f[r] + bv;
        if (LEAKY) v = (v > 0.f) ? v : 0.01f * v;
        C[(size_t)mr * Nc + ncol] = v;
      }
    }
  }
}

// Pack row-major f32 W[K,N] into the WMMA B-fragment layout (bf16).
__global__ void k_pack_w(const float* __restrict__ W, unsigned short* __restrict__ P,
                         int Kd, int Nc)
{
  int id = blockIdx.x * blockDim.x + threadIdx.x;
  int total = Kd * Nc;
  if (id >= total) return;
  int e    = id & 15;
  int lane = (id >> 4) & 31;
  int tk   = id >> 9;
  int KT   = Kd >> 5;
  int kt   = tk % KT, nt = tk / KT;
  int k = kt * 32 + (lane >> 4) * 16 + e;
  int n = nt * 16 + (lane & 15);
  P[id] = f2bf(W[(size_t)k * Nc + n]);
}

// ---------------- feature prep ----------------
__global__ void k_colminmax(const float* __restrict__ A, float* mn, float* mx,
                            int rows, int cols)
{
  int c = blockIdx.x * blockDim.x + threadIdx.x;
  if (c >= cols) return;
  float lo = 3.4e38f, hi = -3.4e38f;
  for (int r = 0; r < rows; ++r) {
    float v = A[(size_t)r * cols + c];
    lo = fminf(lo, v); hi = fmaxf(hi, v);
  }
  mn[c] = lo; mx[c] = hi;
}

__global__ void k_norm_bf16(const float* __restrict__ A, const float* __restrict__ mn,
                            const float* __restrict__ mx, unsigned short* __restrict__ O,
                            int total, int cols)
{
  int id = blockIdx.x * blockDim.x + threadIdx.x;
  if (id >= total) return;
  int c = id % cols;
  O[id] = f2bf((A[id] - mn[c]) / (mx[c] - mn[c]));
}

__global__ void k_norm_f32(float* __restrict__ A, const float* __restrict__ mn,
                           const float* __restrict__ mx, int total, int cols)
{
  int id = blockIdx.x * blockDim.x + threadIdx.x;
  if (id >= total) return;
  int c = id % cols;
  A[id] = (A[id] - mn[c]) / (mx[c] - mn[c]);
}

// sx = concat(emb[res], node_features @ Ws + bs)  -> [N,512]
__global__ void k_build_sx(const int* __restrict__ res, const float* __restrict__ nf,
                           const float* __restrict__ emb, const float* __restrict__ Ws,
                           const float* __restrict__ bs, float* __restrict__ sx, int n)
{
  int id = blockIdx.x * blockDim.x + threadIdx.x;
  if (id >= n * 512) return;
  int r = id >> 9, c = id & 511;
  float v;
  if (c < 128) {
    v = emb[res[r] * 128 + c];
  } else {
    int co = c - 128;
    float acc = bs[co];
    const float* x = nf + (size_t)r * 61;
    for (int k = 0; k < 61; ++k) acc += x[k] * Ws[k * 384 + co];
    v = acc;
  }
  sx[id] = v;
}

// ---------------- graph plumbing ----------------
__global__ void k_degree(const int* __restrict__ src, const int* __restrict__ dst,
                         float* dO, float* dI, int E)
{
  int e = blockIdx.x * blockDim.x + threadIdx.x;
  if (e >= E) return;
  atomicAdd(&dO[src[e]], 1.f);
  atomicAdd(&dI[dst[e]], 1.f);
}

__global__ void k_invsqrt(float* d, int n)
{
  int i = blockIdx.x * blockDim.x + threadIdx.x;
  if (i < n) d[i] = rsqrtf(fmaxf(d[i], 1.f));
}

// agg[dst] += x[src] * no[src] * (ew?)   F fixed at 512 (128 float4 per edge)
template<int WITH_EW>
__global__ void k_scatter(const int* __restrict__ src, const int* __restrict__ dst,
                          const float* __restrict__ no, const float* __restrict__ ew,
                          const float* __restrict__ x, float* __restrict__ agg, int E)
{
  int id = blockIdx.x * blockDim.x + threadIdx.x;
  if (id >= E * 128) return;
  int e = id >> 7;
  int f = (id & 127) << 2;
  int s = src[e], d = dst[e];
  float sc = no[s];
  if (WITH_EW) sc *= ew[e];
  const float4 xv = *(const float4*)(x + (size_t)s * 512 + f);
  float* ap = agg + (size_t)d * 512 + f;
  atomicAdd(ap + 0, xv.x * sc);
  atomicAdd(ap + 1, xv.y * sc);
  atomicAdd(ap + 2, xv.z * sc);
  atomicAdd(ap + 3, xv.w * sc);
}

// out = bf16(agg * ni[row]),  F fixed at 512
__global__ void k_scale_bf16(const float* __restrict__ a, const float* __restrict__ ni,
                             unsigned short* __restrict__ o, int rows)
{
  int id = blockIdx.x * blockDim.x + threadIdx.x;
  if (id >= rows * 512) return;
  o[id] = f2bf(a[id] * ni[id >> 9]);
}

__global__ void k_f32_to_bf16(const float* __restrict__ x, unsigned short* __restrict__ y,
                              int n)
{
  int i = blockIdx.x * blockDim.x + threadIdx.x;
  if (i < n) y[i] = f2bf(x[i]);
}

// ---------------- attention / softmax / recover ----------------
__global__ void k_logits(const float* __restrict__ X, const float* __restrict__ Wg,
                         const float* __restrict__ bg, float* __restrict__ z, int n)
{
  int r = blockIdx.x * blockDim.x + threadIdx.x;
  if (r >= n) return;
  const float4* x = (const float4*)(X + (size_t)r * 256);
  const float4* w = (const float4*)Wg;
  float acc = bg[0];
  for (int i = 0; i < 64; ++i) {
    float4 a = x[i], b = w[i];
    acc += a.x * b.x + a.y * b.y + a.z * b.z + a.w * b.w;
  }
  z[r] = acc;
}

__global__ void k_rmax(const float* __restrict__ z, int n, float* out)
{
  __shared__ float s[1024];
  float m = -3.4e38f;
  for (int i = threadIdx.x; i < n; i += blockDim.x) m = fmaxf(m, z[i]);
  s[threadIdx.x] = m; __syncthreads();
  for (int o = 512; o > 0; o >>= 1) {
    if ((int)threadIdx.x < o) s[threadIdx.x] = fmaxf(s[threadIdx.x], s[threadIdx.x + o]);
    __syncthreads();
  }
  if (threadIdx.x == 0) out[0] = s[0];
}

__global__ void k_rsumexp(const float* __restrict__ z, int n,
                          const float* __restrict__ m, float* out)
{
  __shared__ float s[1024];
  float mm = m[0];
  float acc = 0.f;
  for (int i = threadIdx.x; i < n; i += blockDim.x) acc += expf(z[i] - mm);
  s[threadIdx.x] = acc; __syncthreads();
  for (int o = 512; o > 0; o >>= 1) {
    if ((int)threadIdx.x < o) s[threadIdx.x] += s[threadIdx.x + o];
    __syncthreads();
  }
  if (threadIdx.x == 0) out[0] = s[0];
}

__global__ void k_softmax(const float* __restrict__ z, const float* __restrict__ m,
                          const float* __restrict__ s, float* __restrict__ o, int n)
{
  int i = blockIdx.x * blockDim.x + threadIdx.x;
  if (i < n) o[i] = expf(z[i] - m[0]) / s[0];
}

__global__ void k_rsum(const float* __restrict__ v, int n, float* out)
{
  __shared__ float s[1024];
  float acc = 0.f;
  for (int i = threadIdx.x; i < n; i += blockDim.x) acc += v[i];
  s[threadIdx.x] = acc; __syncthreads();
  for (int o = 512; o > 0; o >>= 1) {
    if ((int)threadIdx.x < o) s[threadIdx.x] += s[threadIdx.x + o];
    __syncthreads();
  }
  if (threadIdx.x == 0) out[0] = s[0];
}

__global__ void k_divide(const float* __restrict__ v, const float* __restrict__ s,
                         float* __restrict__ o, int n)
{
  int i = blockIdx.x * blockDim.x + threadIdx.x;
  if (i < n) o[i] = v[i] / s[0];
}

// overlap-add recover (K=3): out[r,f] = mean over valid kx[r-j,f], j=0..2
template<int F>
__global__ void k_recover(const float* __restrict__ kx, float* __restrict__ out,
                          int nOut, int nIn)
{
  int id = blockIdx.x * blockDim.x + threadIdx.x;
  if (id >= nOut * F) return;
  int r = (F == 1) ? id : (id / F);
  int f = (F == 1) ? 0 : (id - r * F);
  float acc = 0.f; int cnt = 0;
  for (int j = 0; j < 3; ++j) {
    int i = r - j;
    if (i >= 0 && i < nIn) { acc += kx[(size_t)i * F + f]; ++cnt; }
  }
  out[id] = acc / (float)cnt;
}

__global__ void k_mul(const float* __restrict__ a, const float* __restrict__ b,
                      float* __restrict__ o, int n)
{
  int i = blockIdx.x * blockDim.x + threadIdx.x;
  if (i < n) o[i] = a[i] * b[i];
}

// final: out[r,0..1] = f1[r,:128] @ Wf2[128,2] + bf2
__global__ void k_final(const float* __restrict__ f1, const float* __restrict__ Wf2,
                        const float* __restrict__ bf2, float* __restrict__ out, int n)
{
  int r = blockIdx.x * blockDim.x + threadIdx.x;
  if (r >= n) return;
  float a0 = bf2[0], a1 = bf2[1];
  const float* x = f1 + (size_t)r * 128;
  for (int k = 0; k < 128; ++k) {
    float v = x[k];
    a0 += v * Wf2[k * 2 + 0];
    a1 += v * Wf2[k * 2 + 1];
  }
  out[r * 2 + 0] = a0;
  out[r * 2 + 1] = a1;
}

// =====================================================================
extern "C" void kernel_launch(void* const* d_in, const int* in_sizes, int n_in,
                              void* d_out, int out_size, void* d_ws, size_t ws_size,
                              hipStream_t stream)
{
  const int*   struct_src = (const int*)d_in[0];
  const int*   struct_dst = (const int*)d_in[1];
  const int*   res_array  = (const int*)d_in[2];
  const float* node_feat  = (const float*)d_in[3];
  const float* seq_attr   = (const float*)d_in[4];
  const int*   seq_src    = (const int*)d_in[5];
  const int*   seq_dst    = (const int*)d_in[6];
  const float* seq_ew     = (const float*)d_in[7];
  // d_in[8] kmerseq == arange(NQK): identity gather, not needed
  const float* emb  = (const float*)d_in[9];
  const float* Ws   = (const float*)d_in[10]; const float* bs   = (const float*)d_in[11];
  const float* Wq   = (const float*)d_in[12]; const float* bq   = (const float*)d_in[13];
  const float* Wg1s = (const float*)d_in[14]; const float* bg1s = (const float*)d_in[15];
  const float* Wg2s = (const float*)d_in[16]; const float* bg2s = (const float*)d_in[17];
  const float* Wg1q = (const float*)d_in[18]; const float* bg1q = (const float*)d_in[19];
  const float* Wg2q = (const float*)d_in[20]; const float* bg2q = (const float*)d_in[21];
  const float* Wgs  = (const float*)d_in[22]; const float* bgs  = (const float*)d_in[23];
  const float* Wgq  = (const float*)d_in[24]; const float* bgq  = (const float*)d_in[25];
  const float* W2   = (const float*)d_in[26]; const float* b2   = (const float*)d_in[27];
  const float* Wf1  = (const float*)d_in[28]; const float* bf1  = (const float*)d_in[29];
  const float* Wf2  = (const float*)d_in[30]; const float* bf2  = (const float*)d_in[31];

  float* out_final = (float*)d_out;            // [N,2]
  float* out_sattn = out_final + 2 * NR;       // [N]
  float* out_qattn = out_sattn + NR;           // [N]
  float* out_x     = out_qattn + NR;           // [N,256]

  char* base = (char*)d_ws; size_t off = 0;
  auto alloc = [&](size_t bytes) -> void* {
    void* p = base + off; off += (bytes + 255) & ~(size_t)255; return p;
  };

  // bf16 staging for the big GEMM; region reused as agg (f32) after G1 completes
  unsigned short* seqNb = (unsigned short*)alloc((size_t)NQK * 1280 * 2); // 51.2MB
  float* agg = (float*)seqNb;                                            // 41MB < 51.2MB
  unsigned short* aggnb  = (unsigned short*)alloc((size_t)NR * 512 * 2);
  unsigned short* smallb = (unsigned short*)alloc((size_t)NR * 256 * 2);
  unsigned short* WqP  = (unsigned short*)alloc((size_t)1280 * 512 * 2);
  unsigned short* W1sP = (unsigned short*)alloc((size_t)512 * 512 * 2);
  unsigned short* W2sP = (unsigned short*)alloc((size_t)512 * 256 * 2);
  unsigned short* W1qP = (unsigned short*)alloc((size_t)512 * 512 * 2);
  unsigned short* W2qP = (unsigned short*)alloc((size_t)512 * 256 * 2);
  unsigned short* W2P  = (unsigned short*)alloc((size_t)256 * 256 * 2);
  unsigned short* Wf1P = (unsigned short*)alloc((size_t)256 * 128 * 2);
  float* sx   = (float*)alloc((size_t)NR * 512 * 4);
  float* h1   = sx;                              // reuse: sx dead once layer-1 scatter done
  float* seqx = (float*)alloc((size_t)NQK * 512 * 4);
  float* sx2  = (float*)alloc((size_t)NR * 256 * 4);
  float* qg2  = (float*)alloc((size_t)NQK * 256 * 4);
  float* rec  = (float*)alloc((size_t)NR * 256 * 4);
  float* sA2  = (float*)alloc((size_t)NR * 256 * 4);
  float* qA2  = (float*)alloc((size_t)NR * 256 * 4);
  float* f1   = (float*)alloc((size_t)NR * 128 * 4);
  float* cmin = (float*)alloc(1280 * 4); float* cmax = (float*)alloc(1280 * 4);
  float* smin = (float*)alloc(512 * 4);  float* smax = (float*)alloc(512 * 4);
  float* degS = (float*)alloc((size_t)2 * NR * 4);   // [out-deg^-1/2 | in-deg^-1/2]
  float* degQ = (float*)alloc((size_t)2 * NQK * 4);
  float* zS   = (float*)alloc((size_t)NR * 4);
  float* zQ   = (float*)alloc((size_t)NQK * 4);
  float* attnQ = (float*)alloc((size_t)NQK * 4);
  float* recA  = (float*)alloc((size_t)NR * 4);
  float* red   = (float*)alloc(8 * 4);

  const int B = 256;
  auto cdiv = [](long a, long b) { return (int)((a + b - 1) / b); };

  // ---- degrees (both graphs)
  hipMemsetAsync(degS, 0, (size_t)2 * NR * 4, stream);
  hipMemsetAsync(degQ, 0, (size_t)2 * NQK * 4, stream);
  k_degree<<<cdiv(ESN, B), B, 0, stream>>>(struct_src, struct_dst, degS, degS + NR, ESN);
  k_degree<<<cdiv(EQN, B), B, 0, stream>>>(seq_src, seq_dst, degQ, degQ + NQK, EQN);
  k_invsqrt<<<cdiv(2 * NR, B), B, 0, stream>>>(degS, 2 * NR);
  k_invsqrt<<<cdiv(2 * NQK, B), B, 0, stream>>>(degQ, 2 * NQK);

  // ---- pack all GEMM weights into WMMA B-fragment bf16
  k_pack_w<<<cdiv(1280 * 512, B), B, 0, stream>>>(Wq,   WqP,  1280, 512);
  k_pack_w<<<cdiv(512 * 512, B),  B, 0, stream>>>(Wg1s, W1sP, 512, 512);
  k_pack_w<<<cdiv(512 * 256, B),  B, 0, stream>>>(Wg2s, W2sP, 512, 256);
  k_pack_w<<<cdiv(512 * 512, B),  B, 0, stream>>>(Wg1q, W1qP, 512, 512);
  k_pack_w<<<cdiv(512 * 256, B),  B, 0, stream>>>(Wg2q, W2qP, 512, 256);
  k_pack_w<<<cdiv(256 * 256, B),  B, 0, stream>>>(W2,   W2P,  256, 256);
  k_pack_w<<<cdiv(256 * 128, B),  B, 0, stream>>>(Wf1,  Wf1P, 256, 128);

  // ---- seq branch prep: minmax -> bf16, big GEMM seqx = seqN @ Wq + bq
  k_colminmax<<<cdiv(1280, B), B, 0, stream>>>(seq_attr, cmin, cmax, NQK, 1280);
  k_norm_bf16<<<cdiv((long)NQK * 1280, B), B, 0, stream>>>(seq_attr, cmin, cmax, seqNb,
                                                           NQK * 1280, 1280);
  {
    dim3 g(cdiv(NQK, 64), 512 / 128);
    k_gemm_bf16<0, 1280><<<g, B, 0, stream>>>(seqNb, WqP, bq, seqx, NQK, 512);
  }

  // ---- struct branch prep: build sx, minmax in place (f32)
  k_build_sx<<<cdiv((long)NR * 512, B), B, 0, stream>>>(res_array, node_feat, emb, Ws, bs,
                                                        sx, NR);
  k_colminmax<<<cdiv(512, B), B, 0, stream>>>(sx, smin, smax, NR, 512);
  k_norm_f32<<<cdiv((long)NR * 512, B), B, 0, stream>>>(sx, smin, smax, NR * 512, 512);

  // ---- struct gconv layer 1 (512->512, leaky)   [agg region reuses seqNb: G1 done]
  hipMemsetAsync(agg, 0, (size_t)NR * 512 * 4, stream);
  k_scatter<0><<<cdiv((long)ESN * 128, B), B, 0, stream>>>(struct_src, struct_dst, degS,
                                                           nullptr, sx, agg, ESN);
  k_scale_bf16<<<cdiv((long)NR * 512, B), B, 0, stream>>>(agg, degS + NR, aggnb, NR);
  {
    dim3 g(cdiv(NR, 64), 512 / 128);
    k_gemm_bf16<1, 512><<<g, B, 0, stream>>>(aggnb, W1sP, bg1s, h1, NR, 512);
  }
  // ---- struct gconv layer 2 (512->256, leaky)
  hipMemsetAsync(agg, 0, (size_t)NR * 512 * 4, stream);
  k_scatter<0><<<cdiv((long)ESN * 128, B), B, 0, stream>>>(struct_src, struct_dst, degS,
                                                           nullptr, h1, agg, ESN);
  k_scale_bf16<<<cdiv((long)NR * 512, B), B, 0, stream>>>(agg, degS + NR, aggnb, NR);
  {
    dim3 g(cdiv(NR, 64), 256 / 128);
    k_gemm_bf16<1, 512><<<g, B, 0, stream>>>(aggnb, W2sP, bg2s, sx2, NR, 256);
  }

  // ---- struct attention softmax -> out_sattn
  k_logits<<<cdiv(NR, B), B, 0, stream>>>(sx2, Wgs, bgs, zS, NR);
  k_rmax<<<1, 1024, 0, stream>>>(zS, NR, red + 0);
  k_rsumexp<<<1, 1024, 0, stream>>>(zS, NR, red + 0, red + 1);
  k_softmax<<<cdiv(NR, B), B, 0, stream>>>(zS, red + 0, red + 1, out_sattn, NR);

  // ---- seq gconv layer 1 (512->512, leaky, edge-weighted)
  hipMemsetAsync(agg, 0, (size_t)NQK * 512 * 4, stream);
  k_scatter<1><<<cdiv((long)EQN * 128, B), B, 0, stream>>>(seq_src, seq_dst, degQ, seq_ew,
                                                           seqx, agg, EQN);
  k_scale_bf16<<<cdiv((long)NQK * 512, B), B, 0, stream>>>(agg, degQ + NQK, aggnb, NQK);
  {
    dim3 g(cdiv(NQK, 64), 512 / 128);
    k_gemm_bf16<1, 512><<<g, B, 0, stream>>>(aggnb, W1qP, bg1q, h1, NQK, 512);
  }
  // ---- seq gconv layer 2 (512->256, leaky, edge-weighted)
  hipMemsetAsync(agg, 0, (size_t)NQK * 512 * 4, stream);
  k_scatter<1><<<cdiv((long)EQN * 128, B), B, 0, stream>>>(seq_src, seq_dst, degQ, seq_ew,
                                                           h1, agg, EQN);
  k_scale_bf16<<<cdiv((long)NQK * 512, B), B, 0, stream>>>(agg, degQ + NQK, aggnb, NQK);
  {
    dim3 g(cdiv(NQK, 64), 256 / 128);
    k_gemm_bf16<1, 512><<<g, B, 0, stream>>>(aggnb, W2qP, bg2q, qg2, NQK, 256);
  }

  // ---- seq attention softmax, recover, renormalize -> out_qattn
  k_logits<<<cdiv(NQK, B), B, 0, stream>>>(qg2, Wgq, bgq, zQ, NQK);
  k_rmax<<<1, 1024, 0, stream>>>(zQ, NQK, red + 2);
  k_rsumexp<<<1, 1024, 0, stream>>>(zQ, NQK, red + 2, red + 3);
  k_softmax<<<cdiv(NQK, B), B, 0, stream>>>(zQ, red + 2, red + 3, attnQ, NQK);
  k_recover<1><<<cdiv(NR, B), B, 0, stream>>>(attnQ, recA, NR, NQK);
  k_rsum<<<1, 1024, 0, stream>>>(recA, NR, red + 4);
  k_divide<<<cdiv(NR, B), B, 0, stream>>>(recA, red + 4, out_qattn, NR);

  // ---- recover seq features to residues
  k_recover<256><<<cdiv((long)NR * 256, B), B, 0, stream>>>(qg2, rec, NR, NQK);

  // ---- head: sA2 = sx2@W2+b2 ; qA2 = rec@W2+b2 ; x = sA2*qA2 (output)
  k_f32_to_bf16<<<cdiv((long)NR * 256, B), B, 0, stream>>>(sx2, smallb, NR * 256);
  {
    dim3 g(cdiv(NR, 64), 256 / 128);
    k_gemm_bf16<0, 256><<<g, B, 0, stream>>>(smallb, W2P, b2, sA2, NR, 256);
  }
  k_f32_to_bf16<<<cdiv((long)NR * 256, B), B, 0, stream>>>(rec, smallb, NR * 256);
  {
    dim3 g(cdiv(NR, 64), 256 / 128);
    k_gemm_bf16<0, 256><<<g, B, 0, stream>>>(smallb, W2P, b2, qA2, NR, 256);
  }
  k_mul<<<cdiv((long)NR * 256, B), B, 0, stream>>>(sA2, qA2, out_x, NR * 256);

  // ---- f1 = leaky(x@Wf1+bf1) ; final = f1@Wf2+bf2
  k_f32_to_bf16<<<cdiv((long)NR * 256, B), B, 0, stream>>>(out_x, smallb, NR * 256);
  {
    dim3 g(cdiv(NR, 64), 1);   // Ncols=128 handled by the two N-waves of one block column
    k_gemm_bf16<1, 256><<<g, B, 0, stream>>>(smallb, Wf1P, bf1, f1, NR, 128);
  }
  k_final<<<cdiv(NR, B), B, 0, stream>>>(f1, Wf2, bf2, out_final, NR);

  (void)in_sizes; (void)n_in; (void)out_size; (void)ws_size;
}
